// ManifoldReader_14035953123832
// MI455X (gfx1250) — matrix operand order repaired and checked by reading, not compiled, not run on hardware
//
#include <hip/hip_runtime.h>
#include <math.h>
#include <stdint.h>

typedef __attribute__((ext_vector_type(16))) _Float16 v16h;
typedef __attribute__((ext_vector_type(8)))  float    v8f;
typedef __attribute__((ext_vector_type(4)))  int      v4i;

#if defined(__AMDGCN__) && defined(__has_builtin)
#  if __has_builtin(__builtin_amdgcn_global_load_async_to_lds_b128) && \
      __has_builtin(__builtin_amdgcn_s_wait_asynccnt)
#    define USE_ASYNC_LDS 1
#  endif
#endif
#ifndef USE_ASYNC_LDS
#  define USE_ASYNC_LDS 0
#endif

namespace {
constexpr int M = 4, B = 2048, K = 32, D = 256, DZ = 128;
constexpr float SCORE_SCALE = 0.0625f;   // 1/sqrt(256)
constexpr float EPSV = 1e-6f;

// 16-bit A-matrix 16x32 fragment K index (ISA 7.12.2):
// VGPR v<4: lanes0-15 K=2v,2v+1 ; lanes16-31 +8.  v>=4: K=16+2(v-4)+j ; +8.
__device__ __forceinline__ int a_kidx(int e, int half) {
  int v = e >> 1, j = e & 1;
  int base = (v < 4) ? (2 * v + j) : (16 + 2 * (v - 4) + j);
  return base + 8 * half;
}
} // namespace

// ---------------- Kernel A1: Q = z_anc @ W_Q   [B,DZ]x[DZ,D] ----------------
__global__ void __launch_bounds__(256) qproj_kernel(
    const float* __restrict__ z, const float* __restrict__ Wq,
    float* __restrict__ Q) {
  int b = blockIdx.x;        // one row per block
  int e = threadIdx.x;       // 256 = D
  const float* zr = z + (size_t)b * DZ;
  float acc = 0.f;
  #pragma unroll 8
  for (int f = 0; f < DZ; ++f) acc += zr[f] * Wq[f * D + e];
  Q[(size_t)b * D + e] = acc;
}

// ------------- Kernel A2: Qk[m] = Q @ W_K[m]^T  (WMMA f16->f32) -------------
__global__ void __launch_bounds__(256) qk_wmma_kernel(
    const float* __restrict__ Q, const float* __restrict__ Wk,
    float* __restrict__ Qk) {
  int wave = threadIdx.x >> 5;
  int lane = threadIdx.x & 31;
  int tile = blockIdx.x * 8 + wave;              // M*(B/16)*(D/16) tiles
  int m  = tile / ((B / 16) * (D / 16));
  int r  = tile % ((B / 16) * (D / 16));
  int bt = (r / (D / 16)) * 16;
  int dt = (r % (D / 16)) * 16;
  int row  = bt + (lane & 15);
  int half = lane >> 4;
  const float* Wkm = Wk + (size_t)m * D * D;

  v8f acc = {};
  for (int k0 = 0; k0 < D; k0 += 32) {           // reduce over e (8 steps)
    v16h a, bf;
    #pragma unroll
    for (int e = 0; e < 16; ++e)
      a[e] = (_Float16)Q[(size_t)row * D + k0 + a_kidx(e, half)];
    int ncol = dt + (lane & 15);                 // B frag: lane&15 = n, elem = k (table order)
    #pragma unroll
    for (int s = 0; s < 16; ++s)
      bf[s] = (_Float16)Wkm[(size_t)ncol * D + k0 + a_kidx(s, half)];  // W_K[m][d][e] (B^T)
    acc = __builtin_amdgcn_wmma_f32_16x16x32_f16(false, a, false, bf,
                                                 (short)0, acc, false, false);
  }
  // C/D layout: VGPR r -> M=r (lanes<16) / M=r+8 (lanes>=16); lane&15 -> N
  #pragma unroll
  for (int rr = 0; rr < 8; ++rr) {
    int orow = bt + rr + (half << 3);
    int ocol = dt + (lane & 15);
    Qk[((size_t)m * B + orow) * D + ocol] = acc[rr];
  }
}

// --------- Kernel B: streaming attention (HBM-bound, reads h & lbl once) ---------
__global__ void __launch_bounds__(256) stream_attn_kernel(
    const float* __restrict__ h_nei, const float* __restrict__ ew_anc,
    const float* __restrict__ view_mask, const float* __restrict__ lbl_delta,
    const float* __restrict__ lbl_weight, const float* __restrict__ Qk,
    float* __restrict__ Hbar, float* __restrict__ Lbar) {
  __shared__ float h_s[K * D];   // 32 KB tile of h_nei[m,b,:,:]
  __shared__ float q_s[D];
  __shared__ float sc[K];
  __shared__ float at[K];

  int bid = blockIdx.x;          // bid = m*B + b (contiguous h streaming)
  int m = bid / B, b = bid % B;
  int t = threadIdx.x;

  // ---- stage the 32x256 f32 h tile into LDS ----
#if USE_ASYNC_LDS
  {
    const char* gp = (const char*)(h_nei + (size_t)bid * (K * D));
    char* lp0 = (char*)h_s;
    #pragma unroll
    for (int i = 0; i < 8; ++i) {
      size_t off = (size_t)(i * 256 + t) * 16;   // 16B per lane per iter
      __builtin_amdgcn_global_load_async_to_lds_b128(
          (__attribute__((address_space(1))) v4i*)(gp + off),
          (__attribute__((address_space(3))) v4i*)(lp0 + off), 0, 0);
    }
  }
#else
  {
    const float4* hp = (const float4*)(h_nei + (size_t)bid * (K * D));
    float4* hs4 = (float4*)h_s;
    #pragma unroll
    for (int i = 0; i < 8; ++i) hs4[i * 256 + t] = hp[i * 256 + t];
  }
#endif
  q_s[t] = Qk[(size_t)bid * D + t];

  // prefetch the lbl_delta tile now; it's consumed only after softmax
  const float* lp = lbl_delta + ((size_t)b * K * M + m) * D;  // + k*M*D per k
  #pragma unroll
  for (int k = 0; k < K; ++k)
    __builtin_prefetch(lp + (size_t)k * M * D + t, 0, 1);

#if USE_ASYNC_LDS
  __builtin_amdgcn_s_wait_asynccnt(0);
#endif
  __syncthreads();

  // scores: wave w handles k = 4w..4w+3, lane-strided dot + shuffle reduce
  int wave = t >> 5, lane = t & 31;
  for (int kk = 0; kk < 4; ++kk) {
    int k = wave * 4 + kk;
    float p = 0.f;
    #pragma unroll
    for (int i = 0; i < 8; ++i) {
      int d = lane + 32 * i;
      p += h_s[k * D + d] * q_s[d];
    }
    #pragma unroll
    for (int off = 16; off > 0; off >>= 1) p += __shfl_xor(p, off, 32);
    if (lane == 0) {
      int gi = (b * K + k) * M + m;              // [B,K,M] tensors
      float s = p * SCORE_SCALE + __logf(ew_anc[gi] + EPSV) + lbl_weight[gi];
      if (view_mask[gi] == 0.f) s = -1e9f;
      sc[k] = s;
    }
  }
  __syncthreads();

  // stable softmax over K=32 (redundant per thread, broadcast LDS reads)
  float smax = -1e30f;
  #pragma unroll
  for (int k = 0; k < K; ++k) smax = fmaxf(smax, sc[k]);
  float ssum = 0.f;
  #pragma unroll
  for (int k = 0; k < K; ++k) ssum += __expf(sc[k] - smax);
  if (t < K) at[t] = __expf(sc[t] - smax) / ssum;
  __syncthreads();

  // attn-weighted sums: thread t owns column d=t
  float hb = 0.f, lb = 0.f;
  #pragma unroll 4
  for (int k = 0; k < K; ++k) {
    float a = at[k];
    hb += a * h_s[k * D + t];
    lb += a * __builtin_nontemporal_load(lp + (size_t)k * M * D + t);
  }
  Hbar[(size_t)bid * D + t] = hb;
  Lbar[(size_t)bid * D + t] = lb;
}

// ---- Kernel C: out = sum_m vw[m]*(Hbar[m] @ W_V[m]) + sum_m vw[m]*Lbar[m] ----
__global__ void __launch_bounds__(256) out_wmma_kernel(
    const float* __restrict__ Hbar, const float* __restrict__ Lbar,
    const float* __restrict__ Wv, const float* __restrict__ log_vw,
    float* __restrict__ out) {
  int wave = threadIdx.x >> 5, lane = threadIdx.x & 31;
  int tile = blockIdx.x * 8 + wave;              // (B/16)*(D/16) tiles
  int bt = (tile / (D / 16)) * 16;
  int dt = (tile % (D / 16)) * 16;

  // vw = softmax(log_vw), M=4
  float lv[M], vmax = -1e30f;
  #pragma unroll
  for (int i = 0; i < M; ++i) { lv[i] = log_vw[i]; vmax = fmaxf(vmax, lv[i]); }
  float vsum = 0.f;
  #pragma unroll
  for (int i = 0; i < M; ++i) { lv[i] = __expf(lv[i] - vmax); vsum += lv[i]; }
  float vw[M];
  #pragma unroll
  for (int i = 0; i < M; ++i) vw[i] = lv[i] / vsum;

  int row  = bt + (lane & 15);
  int half = lane >> 4;
  v8f acc = {};
  for (int m = 0; m < M; ++m) {
    const float* Hm  = Hbar + (size_t)m * B * D;
    const float* Wvm = Wv + (size_t)m * D * D;
    float s = vw[m];                              // fold view weight into A
    for (int k0 = 0; k0 < D; k0 += 32) {
      v16h a, bf;
      #pragma unroll
      for (int e = 0; e < 16; ++e)
        a[e] = (_Float16)(s * Hm[(size_t)row * D + k0 + a_kidx(e, half)]);
      int ncol = dt + (lane & 15);                // W_V[m][e][d]: row-major row = k
      #pragma unroll
      for (int q = 0; q < 16; ++q)
        bf[q] = (_Float16)Wvm[(size_t)(k0 + a_kidx(q, half)) * D + ncol];
      acc = __builtin_amdgcn_wmma_f32_16x16x32_f16(false, a, false, bf,
                                                   (short)0, acc, false, false);
    }
  }
  #pragma unroll
  for (int rr = 0; rr < 8; ++rr) {
    int orow = bt + rr + (half << 3);
    int ocol = dt + (lane & 15);
    float l = 0.f;
    #pragma unroll
    for (int m = 0; m < M; ++m)
      l += vw[m] * Lbar[((size_t)m * B + orow) * D + ocol];
    out[(size_t)orow * D + ocol] = acc[rr] + l;
  }
}

extern "C" void kernel_launch(void* const* d_in, const int* in_sizes, int n_in,
                              void* d_out, int out_size, void* d_ws, size_t ws_size,
                              hipStream_t stream) {
  (void)in_sizes; (void)n_in; (void)out_size; (void)ws_size;
  const float* h_nei  = (const float*)d_in[0];
  const float* ew_anc = (const float*)d_in[1];
  const float* vmsk   = (const float*)d_in[2];
  const float* z_anc  = (const float*)d_in[3];
  const float* lbl_d  = (const float*)d_in[4];
  const float* lbl_w  = (const float*)d_in[5];
  const float* W_Q    = (const float*)d_in[6];
  const float* W_K    = (const float*)d_in[7];
  const float* W_V    = (const float*)d_in[8];
  const float* log_vw = (const float*)d_in[9];
  float* out = (float*)d_out;

  char* ws = (char*)d_ws;                  // 26 MB used
  size_t qb = (size_t)B * D * sizeof(float);       // 2 MB
  size_t mb = (size_t)M * B * D * sizeof(float);   // 8 MB
  float* Q    = (float*)(ws);
  float* Qk   = (float*)(ws + qb);
  float* Hbar = (float*)(ws + qb + mb);
  float* Lbar = (float*)(ws + qb + 2 * mb);

  qproj_kernel<<<B, D, 0, stream>>>(z_anc, W_Q, Q);
  qk_wmma_kernel<<<(M * (B / 16) * (D / 16)) / 8, 256, 0, stream>>>(Q, W_K, Qk);
  stream_attn_kernel<<<M * B, 256, 0, stream>>>(h_nei, ew_anc, vmsk, lbl_d,
                                                lbl_w, Qk, Hbar, Lbar);
  out_wmma_kernel<<<((B / 16) * (D / 16)) / 8, 256, 0, stream>>>(Hbar, Lbar,
                                                                 W_V, log_vw, out);
}
